// Feature_attentionBlockND_24919400251395
// MI455X (gfx1250) — matrix operand order repaired and checked by reading, not compiled, not run on hardware
//
#include <hip/hip_runtime.h>
#include <hip/hip_bf16.h>

#define B_  4
#define C_  64
#define CI_ 32
#define H_  96
#define W_  96
#define N_  (H_*W_)     // 9216
#define WP_ (W_/2)      // 48
#define M_  ((H_/2)*(W_/2))  // 2304
#define EPS_ 1e-5f

typedef __bf16 bf16;
typedef bf16  v8bf  __attribute__((ext_vector_type(8)));
typedef bf16  v16bf __attribute__((ext_vector_type(16)));
typedef float v8f   __attribute__((ext_vector_type(8)));

// ---- DPP 16-lane butterfly reductions (no LDS, co-executes with WMMA) ----
template<int CTRL>
__device__ __forceinline__ float dpp_bfly(float v) {
    int i = __float_as_int(v);
    i = __builtin_amdgcn_update_dpp(i, i, CTRL, 0xf, 0xf, true);
    return __int_as_float(i);
}
__device__ __forceinline__ float redmax16(float v) {
    v = fmaxf(v, dpp_bfly<0xB1>(v));   // quad_perm lane^1
    v = fmaxf(v, dpp_bfly<0x4E>(v));   // quad_perm lane^2
    v = fmaxf(v, dpp_bfly<0x141>(v));  // row_half_mirror (lane^7 within 8)
    v = fmaxf(v, dpp_bfly<0x140>(v));  // row_mirror (lane^15 within 16)
    return v;
}
__device__ __forceinline__ float redsum16(float v) {
    v += dpp_bfly<0xB1>(v);
    v += dpp_bfly<0x4E>(v);
    v += dpp_bfly<0x141>(v);
    v += dpp_bfly<0x140>(v);
    return v;
}

// ---------------- kernel 0: zero BN accumulators ----------------
__global__ void k_init(float* __restrict__ sums) {
    if (threadIdx.x < 2*CI_) sums[threadIdx.x] = 0.0f;
}

// ---------------- kernel 1: theta projection -> theta[b][n][c] bf16 ----------------
__global__ void k_theta(const float* __restrict__ x, const float* __restrict__ tw,
                        const float* __restrict__ tb, bf16* __restrict__ theta) {
    int idx = blockIdx.x * blockDim.x + threadIdx.x;   // (b, cgroup, n), n fastest
    int n  = idx % N_;
    int t  = idx / N_;
    int cg = t % 4;
    int b  = t / 4;
    const float* xb = x + (size_t)b * C_ * N_;
    float acc[16];
#pragma unroll
    for (int j = 0; j < 16; ++j) acc[j] = tb[cg*16 + j];
    for (int ci = 0; ci < C_; ++ci) {
        float xv = xb[(size_t)ci * N_ + n];
#pragma unroll
        for (int j = 0; j < 16; ++j)
            acc[j] += xv * tw[(cg*16 + j) * C_ + ci];
    }
    bf16* o = theta + ((size_t)b * N_ + n) * C_ + cg*16;
#pragma unroll
    for (int j = 0; j < 16; ++j) o[j] = (bf16)acc[j];
}

// ---------------- kernel 2: g conv + 2x2 maxpool -> gT[b][c][m] bf16 ----------------
__global__ void k_g(const float* __restrict__ x, const float* __restrict__ gw,
                    const float* __restrict__ gbias, bf16* __restrict__ gT) {
    int idx = blockIdx.x * blockDim.x + threadIdx.x;   // (b, c, m), m fastest
    int m = idx % M_;
    int t = idx / M_;
    int c = t % C_;
    int b = t / C_;
    int ph = m / WP_, pw = m % WP_;
    int n00 = (2*ph)*W_ + 2*pw;
    const float* xb = x + (size_t)b * C_ * N_;
    float a0=0.f, a1=0.f, a2=0.f, a3=0.f;
    for (int ci = 0; ci < C_; ++ci) {
        const float* p = xb + (size_t)ci * N_ + n00;
        float w = gw[c * C_ + ci];
        a0 += p[0]*w; a1 += p[1]*w; a2 += p[W_]*w; a3 += p[W_+1]*w;
    }
    float v = fmaxf(fmaxf(a0, a1), fmaxf(a2, a3)) + gbias[c];
    gT[((size_t)b * C_ + c) * M_ + m] = (bf16)v;
}

// ---------------- kernel 3: phi conv + 2x2 maxpool -> phi[b][m][c] bf16 ----------------
__global__ void k_phi(const float* __restrict__ f, const float* __restrict__ pw,
                      const float* __restrict__ pb, bf16* __restrict__ phi) {
    int idx = blockIdx.x * blockDim.x + threadIdx.x;   // (b, m, c), c fastest
    int c = idx % C_;
    int t = idx / C_;
    int m = t % M_;
    int b = t / M_;
    int ph = m / WP_, pwc = m % WP_;
    int n00 = (2*ph)*W_ + 2*pwc;
    const float* fb = f + (size_t)b * CI_ * N_;
    float a0=0.f, a1=0.f, a2=0.f, a3=0.f;
    for (int ci = 0; ci < CI_; ++ci) {
        const float* p = fb + (size_t)ci * N_ + n00;
        float w = pw[c * CI_ + ci];
        a0 += p[0]*w; a1 += p[1]*w; a2 += p[W_]*w; a3 += p[W_+1]*w;
    }
    float v = fmaxf(fmaxf(a0, a1), fmaxf(a2, a3)) + pb[c];
    phi[((size_t)b * M_ + m) * C_ + c] = (bf16)v;
}

// ---------------- kernel 4: fused flash attention + residual ----------------
// grid = B * (N/64), block = 128 (4 waves). Each wave: one 16-row N tile.
__launch_bounds__(128)
__global__ void k_attn(const bf16* __restrict__ theta, const bf16* __restrict__ phi,
                       const bf16* __restrict__ gT, const float* __restrict__ x,
                       float* __restrict__ u) {
    __shared__ __align__(128) bf16 sP[4][16][32];   // per-wave P staging (D->A layout)
    int wave = threadIdx.x >> 5;
    int lane = threadIdx.x & 31;
    int row  = lane & 15;
    int hi   = lane >> 4;
    int b    = blockIdx.x / (N_/64);
    int tile = blockIdx.x % (N_/64);
    int n0   = tile*64 + wave*16;

    const bf16* thb = theta + (size_t)b * N_ * C_;
    const bf16* phb = phi   + (size_t)b * M_ * C_;
    const bf16* gtb = gT    + (size_t)b * C_ * M_;

    // A operand: theta rows n0..n0+15, two K-steps of 32.
    // A k order used here (contiguous, same as B): lane<16 -> K 0..15, lane>=16 -> K 16..31
    v16bf Ath[2];
#pragma unroll
    for (int s = 0; s < 2; ++s) {
        const bf16* p = thb + (size_t)(n0 + row) * C_ + s*32 + hi*16;
        v8bf lo = *(const v8bf*)p;
        v8bf hp = *(const v8bf*)(p + 8);
        Ath[s] = __builtin_shufflevector(lo, hp, 0,1,2,3,4,5,6,7,8,9,10,11,12,13,14,15);
    }

    const v8f z8 = {0.f,0.f,0.f,0.f,0.f,0.f,0.f,0.f};
    v8f acc[4];
#pragma unroll
    for (int t = 0; t < 4; ++t) acc[t] = z8;
    float rmax[8], rsum[8];
#pragma unroll
    for (int r = 0; r < 8; ++r) { rmax[r] = -3.0e38f; rsum[r] = 0.0f; }

    for (int m0 = 0; m0 < M_; m0 += 32) {
        // prefetch next chunk's B operands into cache (global_prefetch_b8)
        if (m0 + 32 < M_) {
            __builtin_prefetch(phb + (size_t)(m0 + 32 + row) * C_ + hi*16, 0, 3);
            __builtin_prefetch(phb + (size_t)(m0 + 48 + row) * C_ + hi*16, 0, 3);
            __builtin_prefetch(gtb + (size_t)row * M_ + m0 + 32 + hi*16, 0, 3);
            __builtin_prefetch(gtb + (size_t)(32 + row) * M_ + m0 + 32 + hi*16, 0, 3);
        }
        // ---- S = theta x phi^T for 32 m-columns (two 16-col tiles, K=64) ----
        v8f S0 = z8, S1 = z8;
#pragma unroll
        for (int s = 0; s < 2; ++s) {
            // B layout: lane<16 holds K 0..15 of column (lane), lane>=16 holds K 16..31
            v16bf B0 = *(const v16bf*)(phb + (size_t)(m0 + row)      * C_ + s*32 + hi*16);
            v16bf B1 = *(const v16bf*)(phb + (size_t)(m0 + 16 + row) * C_ + s*32 + hi*16);
            S0 = __builtin_amdgcn_wmma_f32_16x16x32_bf16(false, Ath[s], false, B0, (short)0, S0, false, false);
            S1 = __builtin_amdgcn_wmma_f32_16x16x32_bf16(false, Ath[s], false, B1, (short)0, S1, false, false);
        }
        // ---- online softmax over this 32-wide chunk ----
#pragma unroll
        for (int r = 0; r < 8; ++r) {
            float s0 = S0[r], s1 = S1[r];
            float cm = redmax16(fmaxf(s0, s1));
            float nm   = fmaxf(rmax[r], cm);
            float corr = __expf(rmax[r] - nm);
            rmax[r] = nm;
            float p0 = __expf(s0 - nm), p1 = __expf(s1 - nm);
            float ps = redsum16(p0 + p1);
            rsum[r] = rsum[r]*corr + ps;
#pragma unroll
            for (int t = 0; t < 4; ++t) acc[t][r] *= corr;
            // D layout: output row = r + 8*hi, column = lane&15
            sP[wave][r + 8*hi][row]      = (bf16)p0;
            sP[wave][r + 8*hi][16 + row] = (bf16)p1;
        }
        asm volatile("" ::: "memory");
        asm volatile("s_wait_dscnt 0" ::: "memory");
        // ---- reload P in A layout ----
        const bf16* pp = &sP[wave][row][hi*16];
        v8bf plo = *(const v8bf*)pp;
        v8bf phi8 = *(const v8bf*)(pp + 8);
        v16bf Ap = __builtin_shufflevector(plo, phi8, 0,1,2,3,4,5,6,7,8,9,10,11,12,13,14,15);
        asm volatile("" ::: "memory");
        // ---- acc += P x G (4 column tiles of 16 channels) ----
#pragma unroll
        for (int ct = 0; ct < 4; ++ct) {
            v16bf Bg = *(const v16bf*)(gtb + (size_t)(ct*16 + row) * M_ + m0 + hi*16);
            acc[ct] = __builtin_amdgcn_wmma_f32_16x16x32_bf16(false, Ap, false, Bg, (short)0, acc[ct], false, false);
        }
    }

    // ---- epilogue: y = acc / rowsum, u = y + x ----
    const float* xb = x + (size_t)b * C_ * N_;
#pragma unroll
    for (int r = 0; r < 8; ++r) {
        int   nrow = n0 + r + 8*hi;
        float inv  = 1.0f / rsum[r];
#pragma unroll
        for (int ct = 0; ct < 4; ++ct) {
            int c = ct*16 + row;
            float yv = acc[ct][r] * inv;
            u[((size_t)b * N_ + nrow) * C_ + c] = yv + xb[(size_t)c * N_ + nrow];
        }
    }
}

// ---------------- kernel 5: W conv (C->CI) + batch-stat partial sums ----------------
__launch_bounds__(256)
__global__ void k_wconv(const float* __restrict__ u, const float* __restrict__ Ww,
                        const float* __restrict__ Wb, float* __restrict__ zpre,
                        float* __restrict__ sums) {
    __shared__ float ssum[CI_], ssq[CI_];
    if (threadIdx.x < CI_) { ssum[threadIdx.x] = 0.f; ssq[threadIdx.x] = 0.f; }
    __syncthreads();
    int idx = blockIdx.x * blockDim.x + threadIdx.x;   // (b, n)
    int n = idx % N_;
    int b = idx / N_;
    float ur[C_];
    const float4* up = (const float4*)(u + ((size_t)b * N_ + n) * C_);
#pragma unroll
    for (int q = 0; q < C_/4; ++q) {
        float4 v = up[q];
        ur[4*q+0]=v.x; ur[4*q+1]=v.y; ur[4*q+2]=v.z; ur[4*q+3]=v.w;
    }
    for (int ci = 0; ci < CI_; ++ci) {
        float a = Wb[ci];
#pragma unroll
        for (int c = 0; c < C_; ++c) a += ur[c] * Ww[ci*C_ + c];
        zpre[((size_t)b*CI_ + ci) * N_ + n] = a;
        atomicAdd(&ssum[ci], a);
        atomicAdd(&ssq[ci],  a*a);
    }
    __syncthreads();
    if (threadIdx.x < CI_) {
        atomicAdd(&sums[threadIdx.x],        ssum[threadIdx.x]);
        atomicAdd(&sums[CI_ + threadIdx.x],  ssq[threadIdx.x]);
    }
}

// ---------------- kernel 6: batchnorm (batch stats) + affine ----------------
__global__ void k_bn(const float* __restrict__ zpre, const float* __restrict__ sums,
                     const float* __restrict__ gamma, const float* __restrict__ beta,
                     float* __restrict__ out) {
    int i  = blockIdx.x * blockDim.x + threadIdx.x;
    int ci = (i / N_) % CI_;
    const float cnt = (float)(B_ * N_);
    float mean = sums[ci] / cnt;
    float var  = sums[CI_ + ci] / cnt - mean*mean;
    float z = zpre[i];
    out[i] = (z - mean) * rsqrtf(var + EPS_) * gamma[ci] + beta[ci];
}

extern "C" void kernel_launch(void* const* d_in, const int* in_sizes, int n_in,
                              void* d_out, int out_size, void* d_ws, size_t ws_size,
                              hipStream_t stream) {
    const float* x    = (const float*)d_in[0];
    const float* feat = (const float*)d_in[1];
    const float* g_w  = (const float*)d_in[2];
    const float* g_b  = (const float*)d_in[3];
    const float* t_w  = (const float*)d_in[4];
    const float* t_b  = (const float*)d_in[5];
    const float* p_w  = (const float*)d_in[6];
    const float* p_b  = (const float*)d_in[7];
    const float* W_w  = (const float*)d_in[8];
    const float* W_b  = (const float*)d_in[9];
    const float* bg   = (const float*)d_in[10];
    const float* bb   = (const float*)d_in[11];
    float* out = (float*)d_out;

    char* ws = (char*)d_ws;
    bf16*  theta = (bf16*)(ws);                     // B*N*C  bf16 = 4,718,592 B
    bf16*  phi   = (bf16*)(ws + 4718592);           // B*M*C  bf16 = 1,179,648 B
    bf16*  gT    = (bf16*)(ws + 5898240);           // B*C*M  bf16 = 1,179,648 B
    float* u     = (float*)(ws + 7077888);          // B*N*C  f32  = 9,437,184 B
    float* zpre  = (float*)(ws + 16515072);         // B*CI*N f32  = 4,718,592 B
    float* sums  = (float*)(ws + 21233664);         // 64 f32

    k_init <<<1, 64, 0, stream>>>(sums);
    k_theta<<<(B_*4*N_)/256,  256, 0, stream>>>(x, t_w, t_b, theta);
    k_g    <<<(B_*C_*M_)/256, 256, 0, stream>>>(x, g_w, g_b, gT);
    k_phi  <<<(B_*M_*C_)/256, 256, 0, stream>>>(feat, p_w, p_b, phi);
    k_attn <<<B_*(N_/64),     128, 0, stream>>>(theta, phi, gT, x, u);
    k_wconv<<<(B_*N_)/256,    256, 0, stream>>>(u, W_w, W_b, zpre, sums);
    k_bn   <<<(B_*CI_*N_)/256,256, 0, stream>>>(zpre, sums, bg, bb, out);
}
